// SelfAttentionLayer_33337536152360
// MI455X (gfx1250) — matrix operand order repaired and checked by reading, not compiled, not run on hardware
//
#include <hip/hip_runtime.h>

typedef __attribute__((ext_vector_type(16))) _Float16 v16h;
typedef __attribute__((ext_vector_type(8)))  _Float16 v8h;
typedef __attribute__((ext_vector_type(8)))  float    v8f;

#define B_  64
#define N_  1024
#define C_  128
#define D_  64
#define WV  4   // waves per block

union AF { v16h v; v8h h[2]; };
union BP { v16h v; unsigned u[8]; };

// A-fragment (16-bit, 16xK): per lane two contiguous 8-half chunks at p and p+16
static __device__ __forceinline__ v16h load_af16(const _Float16* p) {
    AF a; a.h[0] = *(const v8h*)p; a.h[1] = *(const v8h*)(p + 16); return a.v;
}
static __device__ __forceinline__ unsigned pkh2(float a, float b) {
    union { _Float16 h[2]; unsigned u; } t;
    t.h[0] = (_Float16)a; t.h[1] = (_Float16)b; return t.u;
}
#define WMMA16(A,Bm,Cc) __builtin_amdgcn_wmma_f32_16x16x32_f16(false, (A), false, (Bm), (short)0, (Cc), false, false)

// ---------------- Kernel 0: W[C][D] -> WT f16 [3][D][C] ----------------
__global__ __launch_bounds__(256) void wt_kernel(
    const float* __restrict__ Wq, const float* __restrict__ Wk, const float* __restrict__ Wv,
    _Float16* __restrict__ wt)
{
    const float* W = (blockIdx.x == 0) ? Wq : (blockIdx.x == 1) ? Wk : Wv;
    _Float16* o = wt + (size_t)blockIdx.x * D_ * C_;
    for (int i = threadIdx.x; i < D_ * C_; i += 256) {
        const int d = i >> 7, c = i & (C_ - 1);
        o[i] = (_Float16)W[c * D_ + d];
    }
}

// ---------------- Kernel 1: WMMA QKV projection ----------------
// q,k via WT @ x^T  (C-layout: lane m fixed, d varies -> row-major q/k stores)
// v   via x @ Wv    (C-layout: lane d fixed, n varies -> directly stores v^T)
__global__ __launch_bounds__(32 * WV) void proj_kernel(
    const float* __restrict__ x, const _Float16* __restrict__ wt,
    const float* __restrict__ bq, const float* __restrict__ bk, const float* __restrict__ bv,
    float* __restrict__ qf, _Float16* __restrict__ kh, _Float16* __restrict__ vth)
{
    const int lane  = threadIdx.x & 31;
    const int wave  = threadIdx.x >> 5;
    const int tile  = blockIdx.x * WV + wave;   // 0 .. B*N/16-1
    const int row0  = tile << 4;                // base row in [B*N]
    const int b     = row0 >> 10;
    const int nloc  = row0 & (N_ - 1);

    const int col   = lane & 15;
    const int half  = lane >> 4;
    const int kbase = half * 8;
    const int kg16  = half * 16;

    const float* xr = x + (size_t)(row0 + col) * C_;

    // x fragments over K=c (4 chunks of 32): B-frag (col=m) and A-frag (row=m)
    v16h xb[4], xa[4];
    #pragma unroll
    for (int kc = 0; kc < 4; ++kc) {
        const float* p = xr + kc * 32;
        #pragma unroll
        for (int j = 0; j < 16; ++j) xb[kc][j] = (_Float16)p[j < 8 ? kbase + j : kbase + 8 + j];
        #pragma unroll
        for (int i = 0; i < 8; ++i) {
            xa[kc][i]     = (_Float16)p[kbase + i];
            xa[kc][i + 8] = (_Float16)p[kbase + 16 + i];
        }
    }

    const _Float16* WqT = wt;
    const _Float16* WkT = wt + D_ * C_;
    const _Float16* WvT = wt + 2 * D_ * C_;

    #pragma unroll
    for (int dt = 0; dt < 4; ++dt) {
        const int drow = dt * 16 + col;          // d row (q/k A) / d col (v B)
        v8f cq = {}, ck = {}, cv = {};
        #pragma unroll
        for (int kc = 0; kc < 4; ++kc) {
            const int off = drow * C_ + kc * 32;
            v16h aq  = load_af16(WqT + off + kbase);
            v16h akf = load_af16(WkT + off + kbase);
            v16h bvf = load_af16(WvT + off + kbase);
            cq = WMMA16(aq,     xb[kc], cq);
            ck = WMMA16(akf,    xb[kc], ck);
            cv = WMMA16(xa[kc], bvf,    cv);
        }
        // q/k epilogue: lane m = col, d = dt*16 + 8*half + r  (r consecutive)
        {
            const int db = dt * 16 + 8 * half;
            float qv[8], kv[8];
            #pragma unroll
            for (int r = 0; r < 8; ++r) {
                qv[r] = fmaxf(cq[r] + bq[db + r], 0.f);
                kv[r] = fmaxf(ck[r] + bk[db + r], 0.f);
            }
            float* qo = qf + (size_t)(row0 + col) * D_ + db;
            *(float4*)(qo)     = make_float4(qv[0], qv[1], qv[2], qv[3]);
            *(float4*)(qo + 4) = make_float4(qv[4], qv[5], qv[6], qv[7]);
            int4 kp; kp.x = (int)pkh2(kv[0], kv[1]); kp.y = (int)pkh2(kv[2], kv[3]);
            kp.z = (int)pkh2(kv[4], kv[5]); kp.w = (int)pkh2(kv[6], kv[7]);
            *(int4*)(kh + (size_t)(row0 + col) * D_ + db) = kp;
        }
        // v epilogue: lane d = drow fixed, n = nloc + 8*half + r (r consecutive)
        {
            const float bvd = bv[drow];
            float vv[8];
            #pragma unroll
            for (int r = 0; r < 8; ++r) vv[r] = fmaxf(cv[r] + bvd, 0.f);
            int4 vp; vp.x = (int)pkh2(vv[0], vv[1]); vp.y = (int)pkh2(vv[2], vv[3]);
            vp.z = (int)pkh2(vv[4], vv[5]); vp.w = (int)pkh2(vv[6], vv[7]);
            *(int4*)(vth + (size_t)b * D_ * N_ + (size_t)drow * N_ + nloc + 8 * half) = vp;
        }
    }
}

// ---------------- Kernel 2: attention, transposed-scores flash loop ----------------
// T = K_tile @ q^T (scores^T), P' = exp(T) packed in-lane into the B-fragment
// out^T = V^T_tile @ P'. Softmax row sums are accumulated
// per-lane in scalar f32 (each lane privately owns half the keys of its query
// column) and combined with a single end-of-loop half-wave exchange.
__global__ __launch_bounds__(32 * WV) void attn_kernel(
    const float* __restrict__ qf, const _Float16* __restrict__ kh,
    const _Float16* __restrict__ vth, float* __restrict__ out)
{
    const int lane = threadIdx.x & 31;
    const int wave = threadIdx.x >> 5;
    const int tile = blockIdx.x * WV + wave;   // 0 .. B*N/16-1
    const int b    = tile >> 6;
    const int m0   = (tile & 63) << 4;

    const int col   = lane & 15;
    const int half  = lane >> 4;
    const int kbase = half * 8;
    const int kg16  = half * 16;

    const float*    qb = qf  + ((size_t)b * N_ + m0) * D_;
    const _Float16* kb = kh  + ((size_t)b * N_) * D_;
    const _Float16* vb = vth + (size_t)b * D_ * N_;

    // q B-fragments (col = m, K = d), 1/sqrt(D) folded in
    v16h qB0, qB1;
    {
        const float* qp = qb + (size_t)col * D_;
        #pragma unroll
        for (int j = 0; j < 16; ++j) {
            qB0[j] = (_Float16)(qp[     (j < 8 ? kbase + j : kbase + 8 + j)] * 0.125f);
            qB1[j] = (_Float16)(qp[32 + (j < 8 ? kbase + j : kbase + 8 + j)] * 0.125f);
        }
    }

    v8f o0 = {}, o1 = {}, o2 = {}, o3 = {};
    float es = 0.f;                            // per-lane partial softmax denominator

    for (int c = 0; c < 32; ++c) {             // 32 chunks of 32 keys
        const int n0 = c * 32;
        if (c + 1 < 32)
            __builtin_prefetch(kb + (size_t)(n0 + 32 + col) * D_, 0, 1);

        // k A-fragments (row = key, K = d)
        const _Float16* kr0 = kb + (size_t)(n0 +      col) * D_;
        const _Float16* kr1 = kb + (size_t)(n0 + 16 + col) * D_;
        v16h a00 = load_af16(kr0 + kbase);
        v16h a01 = load_af16(kr0 + 32 + kbase);
        v16h a10 = load_af16(kr1 + kbase);
        v16h a11 = load_af16(kr1 + 32 + kbase);

        v8f t0 = {}, t1 = {};
        t0 = WMMA16(a00, qB0, t0);
        t0 = WMMA16(a01, qB1, t0);
        t1 = WMMA16(a10, qB0, t1);
        t1 = WMMA16(a11, qB1, t1);

        // P' = exp(T); per-lane partial sums (scalar VALU co-executes with WMMA)
        float e0[8], e1[8];
        #pragma unroll
        for (int j = 0; j < 8; ++j) { e0[j] = __expf(t0[j]); e1[j] = __expf(t1[j]); }
        float s01 = 0.f, s23 = 0.f;
        #pragma unroll
        for (int j = 0; j < 8; j += 2) {
            s01 += e0[j] + e0[j + 1];
            s23 += e1[j] + e1[j + 1];
        }
        es += s01 + s23;

        unsigned p0[4], p1[4];
        #pragma unroll
        for (int j = 0; j < 4; ++j) {
            p0[j] = pkh2(e0[2 * j], e0[2 * j + 1]);
            p1[j] = pkh2(e1[2 * j], e1[2 * j + 1]);
        }
        // each lane's own exp values form the B-fragment (col = m, K = n)
        BP bp;
        #pragma unroll
        for (int j = 0; j < 4; ++j) {
            unsigned send = half ? p0[j] : p1[j];
            unsigned recv = (unsigned)__shfl_xor((int)send, 16);
            bp.u[j]     = p0[j];
            bp.u[4 + j] = p1[j];
        }

        // v^T A-fragments (row = d, K = n)
        const _Float16* vr = vb + n0 + kbase;
        v16h va0 = load_af16(vr + (size_t)( 0 + col) * N_);
        v16h va1 = load_af16(vr + (size_t)(16 + col) * N_);
        v16h va2 = load_af16(vr + (size_t)(32 + col) * N_);
        v16h va3 = load_af16(vr + (size_t)(48 + col) * N_);

        o0 = WMMA16(va0, bp.v, o0);
        o1 = WMMA16(va1, bp.v, o1);
        o2 = WMMA16(va2, bp.v, o2);
        o3 = WMMA16(va3, bp.v, o3);
    }

    // combine the two half-lane partial sums for this query column
    const float l   = es + __shfl_xor(es, 16);
    const float inv = 1.0f / l;

    // finalize: out[m][d] = o^T[d][m] / l[m] + q[m][d]; lane m = col fixed
    float*       op = out + ((size_t)b * N_ + m0 + col) * D_;
    const float* qp = qb + (size_t)col * D_;

#define FINALIZE(ACC, DT)                                                     \
    {                                                                         \
        const int db = (DT) * 16 + 8 * half;                                  \
        float4 lo, hi;                                                        \
        lo.x = ACC[0] * inv + qp[db + 0]; lo.y = ACC[1] * inv + qp[db + 1];   \
        lo.z = ACC[2] * inv + qp[db + 2]; lo.w = ACC[3] * inv + qp[db + 3];   \
        hi.x = ACC[4] * inv + qp[db + 4]; hi.y = ACC[5] * inv + qp[db + 5];   \
        hi.z = ACC[6] * inv + qp[db + 6]; hi.w = ACC[7] * inv + qp[db + 7];   \
        *(float4*)(op + db)     = lo;                                         \
        *(float4*)(op + db + 4) = hi;                                         \
    }
    FINALIZE(o0, 0)
    FINALIZE(o1, 1)
    FINALIZE(o2, 2)
    FINALIZE(o3, 3)
#undef FINALIZE
}

extern "C" void kernel_launch(void* const* d_in, const int* in_sizes, int n_in,
                              void* d_out, int out_size, void* d_ws, size_t ws_size,
                              hipStream_t stream)
{
    const float* x  = (const float*)d_in[0];
    const float* Wq = (const float*)d_in[1];
    const float* bq = (const float*)d_in[2];
    const float* Wk = (const float*)d_in[3];
    const float* bk = (const float*)d_in[4];
    const float* Wv = (const float*)d_in[5];
    const float* bv = (const float*)d_in[6];
    float* out = (float*)d_out;

    char* ws = (char*)d_ws;
    float*    qf  = (float*)ws;                                   // 16 MB: q  f32 [B][N][D]
    _Float16* kh  = (_Float16*)(ws + (size_t)16 * 1024 * 1024);   //  8 MB: k  f16 [B][N][D]
    _Float16* vth = (_Float16*)(ws + (size_t)24 * 1024 * 1024);   //  8 MB: v^T f16 [B][D][N]
    _Float16* wt  = (_Float16*)(ws + (size_t)32 * 1024 * 1024);   // 48 KB: W^T f16 [3][D][C]

    wt_kernel  <<<3, 256, 0, stream>>>(Wq, Wk, Wv, wt);
    proj_kernel<<<(B_ * N_ / 16) / WV, 32 * WV, 0, stream>>>(x, wt, bq, bk, bv, qf, kh, vth);
    attn_kernel<<<(B_ * N_ / 16) / WV, 32 * WV, 0, stream>>>(qf, kh, vth, out);
}
